// CrossAttentionRope_53437983097103
// MI455X (gfx1250) — matrix operand order repaired and checked by reading, not compiled, not run on hardware
//
#include <hip/hip_runtime.h>

// ---------- problem constants ----------
#define NB   16
#define NQ   256
#define NKV  4096
#define NH   12
#define HD   64
#define DIMF 768

typedef __bf16 bf16;
typedef __attribute__((ext_vector_type(16))) __bf16 v16bf;
typedef __attribute__((ext_vector_type(8)))  __bf16 v8bf;
typedef __attribute__((ext_vector_type(8)))  float  v8f;

union AF { v16bf v; v8bf h[2]; };

__device__ inline unsigned short f2bf_u(float f) {
  unsigned u = __builtin_bit_cast(unsigned, f);
  return (unsigned short)((u + 0x7fffu + ((u >> 16) & 1u)) >> 16);
}
__device__ inline bf16 f2bf(float f) {
  unsigned short s = f2bf_u(f);
  union { unsigned short u; bf16 b; } c; c.u = s; return c.b;
}
__device__ inline v8bf ld8(const bf16* p) { return *(const v8bf*)p; }
__device__ inline v8bf ld8s(const unsigned short* p) { return *(const v8bf*)(const void*)p; }

__device__ inline float rmax16(float x) {
#pragma unroll
  for (int m = 1; m < 16; m <<= 1) x = fmaxf(x, __shfl_xor(x, m, 32));
  return x;
}
__device__ inline float rsum16(float x) {
#pragma unroll
  for (int m = 1; m < 16; m <<= 1) x += __shfl_xor(x, m, 32);
  return x;
}

// ---------- stage 1: fp32 -> bf16 ----------
__global__ __launch_bounds__(256) void cvt_kernel(const float* __restrict__ in,
                                                  bf16* __restrict__ out, size_t n) {
  for (size_t i = (size_t)blockIdx.x * blockDim.x + threadIdx.x; i < n;
       i += (size_t)gridDim.x * blockDim.x)
    out[i] = f2bf(in[i]);
}

// W[k][n] fp32 -> Wt[n][k] bf16 (B-fragment friendly: fixed N row, contiguous K)
__global__ __launch_bounds__(256) void twt_kernel(const float* __restrict__ W,
                                                  bf16* __restrict__ Wt) {
  int i = blockIdx.x * 256 + threadIdx.x;            // i over 768*768
  int k = i / DIMF, n = i % DIMF;
  Wt[(size_t)n * DIMF + k] = f2bf(W[i]);
}

// ---------- stage 2: WMMA projection (+ optional RoPE, layout store) ----------
// One wave computes a 32x64 tile: 2 A-frags x 4 B-frags -> 8 wmma per k-step.
// mode 0: Q  -> outb[b,h,q,d]  with rope
// mode 1: K  -> outb[b,h,kv,d] with rope
// mode 2: V  -> outb[b,h,d,kv] (transposed), no rope
// mode 3: O  -> outf[row,col]  fp32, no rope
__global__ __launch_bounds__(256) void proj_kernel(const bf16* __restrict__ X,
                                                   const bf16* __restrict__ Wt,
                                                   const float* __restrict__ rope,
                                                   bf16* __restrict__ outb,
                                                   float* __restrict__ outf,
                                                   int rows, int nper, int mode) {
  const int wid  = (blockIdx.x << 3) + (threadIdx.x >> 5);
  const int lane = threadIdx.x & 31;
  const int grp  = lane >> 4, ln = lane & 15;
  const int mt = wid / 12, ng = wid % 12;            // 12 col groups of 64
  if (mt * 32 >= rows) return;

  const bf16* arow0 = X + (size_t)(mt * 32 + ln) * DIMF;
  const bf16* arow1 = arow0 + (size_t)16 * DIMF;
  v8f z = {0.f,0.f,0.f,0.f,0.f,0.f,0.f,0.f};
  v8f acc[2][4] = {{z, z, z, z}, {z, z, z, z}};

  for (int k = 0; k < DIMF; k += 32) {
    AF a0, a1;                                       // A: two 16x32 bf16 tiles
    a0.h[0] = ld8(arow0 + k + grp * 8);
    a0.h[1] = ld8(arow0 + k + 16 + grp * 8);
    a1.h[0] = ld8(arow1 + k + grp * 8);
    a1.h[1] = ld8(arow1 + k + 16 + grp * 8);
#pragma unroll
    for (int t = 0; t < 4; t++) {
      const bf16* wrow = Wt + (size_t)(ng * 64 + t * 16 + ln) * DIMF + k + grp * 8;
      AF b; b.h[0] = ld8(wrow); b.h[1] = ld8(wrow + 16);
      acc[0][t] = __builtin_amdgcn_wmma_f32_16x16x32_bf16(false, a0.v, false, b.v,
                                                          (short)0, acc[0][t], false, false);
      acc[1][t] = __builtin_amdgcn_wmma_f32_16x16x32_bf16(false, a1.v, false, b.v,
                                                          (short)0, acc[1][t], false, false);
    }
  }

#pragma unroll
  for (int rt = 0; rt < 2; rt++) {
#pragma unroll
    for (int t = 0; t < 4; t++) {
#pragma unroll
      for (int r = 0; r < 8; r++) {
        float val = acc[rt][t][r];
        float par = __shfl_xor(val, 1, 32);          // interleaved RoPE partner
        int Mrow = mt * 32 + rt * 16 + r + grp * 8;
        int col  = ng * 64 + t * 16 + ln;
        if (mode == 3) { outf[(size_t)Mrow * DIMF + col] = val; continue; }
        int bI = Mrow / nper, pos = Mrow % nper;
        int hI = col >> 6, d = col & 63;
        float v2 = val;
        if (mode <= 1) {
          float sn = rope[pos * 2 * HD + d];
          float cs = rope[pos * 2 * HD + HD + d];
          v2 = (d & 1) ? val * cs + par * sn : val * cs - par * sn;
        }
        size_t idx;
        if (mode == 2) idx = ((size_t)(bI * NH + hI) * HD + d) * NKV + pos;
        else           idx = ((size_t)(bI * NH + hI) * nper + pos) * HD + d;
        outb[idx] = f2bf(v2);
      }
    }
  }
}

// ---------- stage 3: flash attention, one wave per (b,h,16 q rows) ----------
__global__ __launch_bounds__(256) void attn_kernel(const bf16* __restrict__ Q,
                                                   const bf16* __restrict__ K,
                                                   const bf16* __restrict__ V,
                                                   bf16* __restrict__ O) {
  __shared__ __attribute__((aligned(16))) unsigned short smem[8][16 * 32];
  const int wIn  = threadIdx.x >> 5;
  const int wid  = (blockIdx.x << 3) + wIn;
  const int lane = threadIdx.x & 31;
  const int grp  = lane >> 4, ln = lane & 15;
  const int qt = wid & 15;                            // NQ/16 = 16 tiles
  const int bh = wid >> 4;
  unsigned short* pl = smem[wIn];

  // Q A-fragments for two k-steps (d 0..31, 32..63)
  const bf16* qrow = Q + ((size_t)bh * NQ + qt * 16 + ln) * HD;
  AF aq[2];
#pragma unroll
  for (int ks = 0; ks < 2; ks++) {
    aq[ks].h[0] = ld8(qrow + ks * 32 + grp * 8);
    aq[ks].h[1] = ld8(qrow + ks * 32 + 16 + grp * 8);
  }

  v8f z = {0.f,0.f,0.f,0.f,0.f,0.f,0.f,0.f};
  v8f o[4] = {z, z, z, z};
  float m[8], l[8];
#pragma unroll
  for (int r = 0; r < 8; r++) { m[r] = -1e30f; l[r] = 0.f; }
  const float scale = 0.125f;                         // 1/sqrt(64)

  for (int kv0 = 0; kv0 < NKV; kv0 += 32) {
    // ---- S = Q * K^T : two 16x16 tiles ----
    v8f s[2];
#pragma unroll
    for (int j = 0; j < 2; j++) {
      v8f cc = z;
#pragma unroll
      for (int ks = 0; ks < 2; ks++) {
        const bf16* kr = K + ((size_t)bh * NKV + kv0 + j * 16 + ln) * HD + ks * 32 + grp * 8;
        AF bk; bk.h[0] = ld8(kr); bk.h[1] = ld8(kr + 16);
        cc = __builtin_amdgcn_wmma_f32_16x16x32_bf16(false, aq[ks].v, false, bk.v,
                                                     (short)0, cc, false, false);
      }
      s[j] = cc;
    }
    if (kv0 + 32 < NKV) {
      __builtin_prefetch(K + ((size_t)bh * NKV + kv0 + 32 + ln) * HD, 0, 1);
      __builtin_prefetch(V + ((size_t)bh * HD + ln) * NKV + kv0 + 32, 0, 1);
    }
    // ---- online softmax (row r lives in lanes 0-15 / r+8 in 16-31) ----
    float f[8];
#pragma unroll
    for (int r = 0; r < 8; r++) {
      float s0 = s[0][r] * scale, s1 = s[1][r] * scale;
      float bm = rmax16(fmaxf(s0, s1));
      float mn = fmaxf(m[r], bm);
      float fr = __expf(m[r] - mn);
      float p0 = __expf(s0 - mn), p1 = __expf(s1 - mn);
      float rs = rsum16(p0 + p1);
      l[r] = l[r] * fr + rs; m[r] = mn; f[r] = fr;
      int row = r + grp * 8;
      pl[row * 32 + ln]      = f2bf_u(p0);           // P tile -> LDS (16x32 bf16)
      pl[row * 32 + 16 + ln] = f2bf_u(p1);
    }
#pragma unroll
    for (int t = 0; t < 4; t++)
#pragma unroll
      for (int r = 0; r < 8; r++) o[t][r] *= f[r];
    asm volatile("s_wait_dscnt 0" ::: "memory");     // intra-wave LDS RAW
    // ---- re-read P as A-fragment, O += P * V ----
    AF pa;
    pa.h[0] = ld8s(pl + ln * 32 + grp * 8);
    pa.h[1] = ld8s(pl + ln * 32 + 16 + grp * 8);
#pragma unroll
    for (int t = 0; t < 4; t++) {
      const bf16* vr = V + ((size_t)bh * HD + t * 16 + ln) * NKV + kv0 + grp * 8;
      AF bv; bv.h[0] = ld8(vr); bv.h[1] = ld8(vr + 16);
      o[t] = __builtin_amdgcn_wmma_f32_16x16x32_bf16(false, pa.v, false, bv.v,
                                                     (short)0, o[t], false, false);
    }
  }

  // epilogue: normalize + store bf16 [b, q, h*64+d]
  const int b = bh / NH, h = bh % NH;
#pragma unroll
  for (int t = 0; t < 4; t++)
#pragma unroll
    for (int r = 0; r < 8; r++) {
      int qr = qt * 16 + r + grp * 8;
      size_t idx = ((size_t)(b * NQ + qr)) * DIMF + h * 64 + t * 16 + ln;
      O[idx] = f2bf(o[t][r] / l[r]);
    }
}

// ---------- host launcher ----------
extern "C" void kernel_launch(void* const* d_in, const int* in_sizes, int n_in,
                              void* d_out, int out_size, void* d_ws, size_t ws_size,
                              hipStream_t stream) {
  const float* x     = (const float*)d_in[0];
  const float* ctx   = (const float*)d_in[1];
  const float* ropeq = (const float*)d_in[2];
  const float* ropek = (const float*)d_in[3];
  const float* Wq    = (const float*)d_in[4];
  const float* Wk    = (const float*)d_in[5];
  const float* Wv    = (const float*)d_in[6];
  const float* Wo    = (const float*)d_in[7];
  float* out = (float*)d_out;

  char* ws = (char*)d_ws;
  const size_t szX  = (size_t)NB * NQ  * DIMF * 2;   // 6.3 MB
  const size_t szC  = (size_t)NB * NKV * DIMF * 2;   // 100.7 MB
  const size_t szW  = (size_t)DIMF * DIMF * 2;       // 1.18 MB
  const size_t szQ  = (size_t)NB * NH * NQ  * HD * 2;
  const size_t szK  = (size_t)NB * NH * NKV * HD * 2;
  size_t off = 0;
  bf16* xb  = (bf16*)(ws + off); off += szX;
  bf16* cb  = (bf16*)(ws + off); off += szC;
  bf16* Wqt = (bf16*)(ws + off); off += szW;
  bf16* Wkt = (bf16*)(ws + off); off += szW;
  bf16* Wvt = (bf16*)(ws + off); off += szW;
  bf16* Wot = (bf16*)(ws + off); off += szW;
  bf16* Qr  = (bf16*)(ws + off); off += szQ;
  bf16* Kr  = (bf16*)(ws + off); off += szK;
  bf16* Vt  = (bf16*)(ws + off); off += szK;
  bf16* Ob  = (bf16*)(ws + off); off += szX;

  // stage 1: convert + transpose weights
  cvt_kernel<<<2048, 256, 0, stream>>>(x,   xb, (size_t)NB * NQ  * DIMF);
  cvt_kernel<<<4096, 256, 0, stream>>>(ctx, cb, (size_t)NB * NKV * DIMF);
  twt_kernel<<<(DIMF * DIMF) / 256, 256, 0, stream>>>(Wq, Wqt);
  twt_kernel<<<(DIMF * DIMF) / 256, 256, 0, stream>>>(Wk, Wkt);
  twt_kernel<<<(DIMF * DIMF) / 256, 256, 0, stream>>>(Wv, Wvt);
  twt_kernel<<<(DIMF * DIMF) / 256, 256, 0, stream>>>(Wo, Wot);

  // stage 2: projections — one wave per 32x64 tile, 8 waves/block
  const int rowsQ = NB * NQ, rowsK = NB * NKV;
  proj_kernel<<<(rowsQ / 32) * 12 / 8, 256, 0, stream>>>(xb, Wqt, ropeq, Qr, nullptr, rowsQ, NQ,  0);
  proj_kernel<<<(rowsK / 32) * 12 / 8, 256, 0, stream>>>(cb, Wkt, ropek, Kr, nullptr, rowsK, NKV, 1);
  proj_kernel<<<(rowsK / 32) * 12 / 8, 256, 0, stream>>>(cb, Wvt, ropek, Vt, nullptr, rowsK, NKV, 2);

  // stage 3: attention — NB*NH*(NQ/16) = 3072 waves
  attn_kernel<<<(NB * NH * (NQ / 16)) / 8, 256, 0, stream>>>(Qr, Kr, Vt, Ob);

  // stage 4: output projection -> fp32 d_out
  proj_kernel<<<(rowsQ / 32) * 12 / 8, 256, 0, stream>>>(Ob, Wot, ropeq, nullptr, out, rowsQ, NQ, 3);
}
